// Block_23244363006479
// MI455X (gfx1250) — compile-verified, matrix operand order checked by reading, not run on hardware
//
#include <hip/hip_runtime.h>
#include <stdint.h>

// ---------------------------------------------------------------------------
// Problem constants (match reference)
// ---------------------------------------------------------------------------
#define BB   4
#define TT   2048
#define DDIM 1024
#define HH   16
#define HDD  64
#define FFD  16384
#define MROWS (BB * TT)   // 8192

typedef __attribute__((ext_vector_type(16))) __bf16    v16bf;
typedef __attribute__((ext_vector_type(8)))  float     v8f;
typedef __attribute__((ext_vector_type(4)))  unsigned  u32x4;
typedef __attribute__((ext_vector_type(4)))  int       i32x4;
typedef __attribute__((ext_vector_type(8)))  int       i32x8;

// Tensor Data Mover availability (probe: ROCm 7.2 clang-22 = 5-arg builtin,
// amdgpu-toolchain clang-23 = 6-arg builtin).
#if defined(__has_builtin)
#if __has_builtin(__builtin_amdgcn_tensor_load_to_lds) && \
    __has_builtin(__builtin_amdgcn_s_wait_tensorcnt)
#define HAVE_TDM 1
#endif
#endif
#ifndef HAVE_TDM
#define HAVE_TDM 0
#endif

// ---------------------------------------------------------------------------
// bf16 helpers (bit-level RNE-ish conversion; no reliance on __bf16 arithmetic)
// ---------------------------------------------------------------------------
__device__ __forceinline__ unsigned short f2bf(float f) {
    unsigned int u = __float_as_uint(f);
    u += 0x7FFFu + ((u >> 16) & 1u);
    return (unsigned short)(u >> 16);
}

union FragBF { v16bf v; u32x4 q[2]; };

// Load one WMMA 16-bit operand fragment (16x32 A-layout / symmetric B-layout)
// from LDS. `rowbase` points at this lane's row; `koff` selects the 32-wide
// k-window (0 or 32); `kh` = lane>>4 selects the k-half per the VGPR layout:
//   lanes 0-15 hold K {0..7, 16..23}, lanes 16-31 hold K {8..15, 24..31}.
__device__ __forceinline__ v16bf load_frag(const unsigned short* rowbase,
                                           int koff, int kh) {
    FragBF f;
    f.q[0] = *(const u32x4*)(rowbase + koff + kh * 8);
    f.q[1] = *(const u32x4*)(rowbase + koff + 16 + kh * 8);
    return f.v;
}

__device__ __forceinline__ v8f wmma_bf16(v16bf a, v16bf b, v8f c) {
    return __builtin_amdgcn_wmma_f32_16x16x32_bf16(
        /*neg_a=*/false, a, /*neg_b=*/false, b,
        /*c_mod=*/(short)0, c, /*reuse_a=*/false, /*reuse_b=*/false);
}

#if HAVE_TDM
// ---------------------------------------------------------------------------
// TDM: DMA one 2D bf16 tile [rows x cols] (cols elems contiguous per row,
// row stride `stride_elems` in global) into LDS, with `pad_dw` DWORDs of LDS
// padding appended after every `cols` elements (cols*2 bytes must be a
// multiple of 4 and match a legal pad_interval: here cols is 32 or 64).
//   cols=32 -> 16 DW rows -> pad_interval code 3 ; cols=64 -> 32 DW -> code 4
// pad_dw=4 -> pad_amount code 3 (amount = code+1 DWORDs).
// D# layout per CDNA5 ISA.
// ---------------------------------------------------------------------------
template <unsigned COLS>
__device__ __forceinline__ void tdm_load_tile(unsigned lds_addr,
                                              const void* gsrc,
                                              unsigned stride_elems,
                                              unsigned rows) {
    const unsigned long long ga = (unsigned long long)(uintptr_t)gsrc;
    u32x4 g0 = { 1u,                                   // count=1, user mode
                 lds_addr,
                 (unsigned)ga,
                 (unsigned)((ga >> 32) & 0x1FFFFFFull) | (2u << 30) }; // type=2
    const unsigned padint = (COLS == 32) ? 3u : 4u;    // 16 or 32 DW interval
    const unsigned td0 = stride_elems;                 // tensor_dim0
    const unsigned td1 = 1u << 20;                     // tensor_dim1 (ample)
    i32x8 g1;
    g1[0] = (int)((1u << 16)            // data_size = 2 bytes
                | (1u << 20)            // pad_enable
                | (padint << 22)        // pad_interval
                | (3u << 25));          // pad_amount = 4 DWORDs
    g1[1] = (int)((td0 & 0xFFFFu) << 16);              // dim0[15:0]
    g1[2] = (int)((td0 >> 16) | ((td1 & 0xFFFFu) << 16));
    g1[3] = (int)((td1 >> 16) | (COLS << 16));         // tile_dim0
    g1[4] = (int)(rows & 0xFFFFu);                     // tile_dim1, tile_dim2=0
    g1[5] = (int)stride_elems;                         // dim0_stride[31:0]
    g1[6] = 0;
    g1[7] = 0;
    i32x4 z4 = { 0, 0, 0, 0 };
#if __clang_major__ >= 23
    i32x8 z8 = { 0, 0, 0, 0, 0, 0, 0, 0 };
    __builtin_amdgcn_tensor_load_to_lds(g0, g1, z4, z4, z8, 0);
#else
    __builtin_amdgcn_tensor_load_to_lds(g0, g1, z4, z4, 0);
#endif
}
#endif  // HAVE_TDM

// ---------------------------------------------------------------------------
// Prep kernels
// ---------------------------------------------------------------------------
__global__ void cvt_bf16_kernel(const float* __restrict__ in,
                                unsigned short* __restrict__ out, size_t n) {
    for (size_t i = (size_t)blockIdx.x * blockDim.x + threadIdx.x; i < n;
         i += (size_t)gridDim.x * blockDim.x)
        out[i] = f2bf(in[i]);
}

// Build Wqkvt[n][dd] (bf16, [3*D][D]) from Wq/Wk/Wv [H, D, HD]:
//   n = which*D + h*HD + d ;  src = W[h][dd][d]
__global__ void pack_qkv_w_kernel(const float* __restrict__ Wq,
                                  const float* __restrict__ Wk,
                                  const float* __restrict__ Wv,
                                  unsigned short* __restrict__ out) {
    const size_t ntot = (size_t)3 * DDIM * DDIM;
    for (size_t i = (size_t)blockIdx.x * blockDim.x + threadIdx.x; i < ntot;
         i += (size_t)gridDim.x * blockDim.x) {
        int dd  = (int)(i & (DDIM - 1));
        int n   = (int)(i >> 10);
        int which = n >> 10;
        int rem   = n & (DDIM - 1);
        int h = rem >> 6, d = rem & (HDD - 1);
        const float* W = (which == 0) ? Wq : (which == 1) ? Wk : Wv;
        out[i] = f2bf(W[((size_t)h * DDIM + dd) * HDD + d]);
    }
}

// out[c][r] (bf16, [C][R]) = in[r][c] (fp32, [R][C])
__global__ void transpose_bf16_kernel(const float* __restrict__ in,
                                      unsigned short* __restrict__ out,
                                      int R, int C) {
    const size_t n = (size_t)R * C;
    for (size_t i = (size_t)blockIdx.x * blockDim.x + threadIdx.x; i < n;
         i += (size_t)gridDim.x * blockDim.x) {
        size_t r = i / (size_t)C, c = i % (size_t)C;
        out[c * (size_t)R + r] = f2bf(in[i]);
    }
}

// ---------------------------------------------------------------------------
// Generic bf16 WMMA GEMM:  C[M,N] = A[M,K] (bf16, row-major)
//                                   x Bt[N,K]^T (bf16, N-major == B[K,N])
// Block tile 128x128, BK=32, 256 threads = 8 waves in a 4(M) x 2(N) grid,
// each wave owns a 32x64 sub-tile = 8 WMMA accumulators.
// Staging: TDM double-buffered (wave 0 drives the DMA; TENSORcnt in-order
// retirement) when available, else global->reg->LDS.
// MODE 0: QKV epilogue   (bias0/1/2 = bq/bk/bv; outH0/1/2 = q/k/v [B,H,T,64];
//                         Q scaled by 1/sqrt(HD))
// MODE 1: out-proj       (+bias0 +resid -> outF fp32)
// MODE 2: FFN1           (+bias0, ReLU -> outH0 bf16)
// MODE 3: FFN2           (+bias0 +resid -> outF fp32)
// ---------------------------------------------------------------------------
template <int MODE>
__global__ __launch_bounds__(256) void gemm_bf16_kernel(
    const unsigned short* __restrict__ A,
    const unsigned short* __restrict__ Bt,
    int Ntot, int Ktot,
    const float* __restrict__ bias0,
    const float* __restrict__ bias1,
    const float* __restrict__ bias2,
    const float* __restrict__ resid,
    float* __restrict__ outF,
    unsigned short* __restrict__ outH0,
    unsigned short* __restrict__ outH1,
    unsigned short* __restrict__ outH2) {
#if HAVE_TDM
    __shared__ __align__(16) unsigned short As[2 * 128 * 40];
    __shared__ __align__(16) unsigned short Bs[2 * 128 * 40];
#else
    __shared__ __align__(16) unsigned short As[128 * 40];
    __shared__ __align__(16) unsigned short Bs[128 * 40];
#endif

    const int tid  = threadIdx.x;
    const int lane = tid & 31;
    const int wave = tid >> 5;
    const int wm = wave >> 1;            // 0..3  (M sub-tile)
    const int wn = wave & 1;             // 0..1  (N sub-tile)
    const int m0 = blockIdx.y * 128;
    const int n0 = blockIdx.x * 128;
    const int kh = lane >> 4;
    const int cl = lane & 15;
    const int nk = Ktot >> 5;

    v8f acc[2][4];
    for (int i = 0; i < 2; ++i)
        for (int j = 0; j < 4; ++j) acc[i][j] = v8f{};

    auto mma_step = [&](const unsigned short* Ab, const unsigned short* Bb) {
        const unsigned short* arow0 = Ab + (wm * 32 + cl) * 40;
        const unsigned short* arow1 = arow0 + 16 * 40;
        const unsigned short* brow0 = Bb + (wn * 64 + cl) * 40;
        v16bf af0 = load_frag(arow0, 0, kh);
        v16bf af1 = load_frag(arow1, 0, kh);
#pragma unroll
        for (int j = 0; j < 4; ++j) {
            v16bf bf = load_frag(brow0 + j * 16 * 40, 0, kh);
            acc[0][j] = wmma_bf16(af0, bf, acc[0][j]);
            acc[1][j] = wmma_bf16(af1, bf, acc[1][j]);
        }
    };

#if HAVE_TDM
    // ---- TDM double-buffered pipeline (wave 0 drives the DMA engine) ----
    const unsigned ldsA = (unsigned)(uintptr_t)As;
    const unsigned ldsB = (unsigned)(uintptr_t)Bs;
    const unsigned TILE_BYTES = 128 * 40 * 2;
    const unsigned short* Abase = A  + (size_t)m0 * Ktot;
    const unsigned short* Bbase = Bt + (size_t)n0 * Ktot;
    if (wave == 0) {
        tdm_load_tile<32>(ldsA, Abase, (unsigned)Ktot, 128);
        tdm_load_tile<32>(ldsB, Bbase, (unsigned)Ktot, 128);
    }
    for (int kt = 0; kt < nk; ++kt) {
        const int cur = kt & 1;
        if (wave == 0) {
            if (kt + 1 < nk) {
                // buf[1-cur] was last read in iteration kt-1; the barrier that
                // closed kt-1 makes it safe to overwrite now.
                const unsigned nxt = (unsigned)(1 - cur) * TILE_BYTES;
                tdm_load_tile<32>(ldsA + nxt, Abase + (size_t)(kt + 1) * 32,
                                  (unsigned)Ktot, 128);
                tdm_load_tile<32>(ldsB + nxt, Bbase + (size_t)(kt + 1) * 32,
                                  (unsigned)Ktot, 128);
                __builtin_amdgcn_s_wait_tensorcnt(2);  // retire pair for kt
            } else {
                __builtin_amdgcn_s_wait_tensorcnt(0);
            }
        }
        __syncthreads();                 // broadcast DMA completion
        mma_step(As + cur * (128 * 40), Bs + cur * (128 * 40));
        __syncthreads();                 // all waves done reading buf[cur]
    }
#else
    // ---- fallback: global->reg->LDS with register prefetch ----
    const int lr  = tid >> 2;            // 0..63
    const int seg = (tid & 3) * 8;       // 0,8,16,24 (elements)
    const unsigned short* Ag0 = A  + (size_t)(m0 + lr)      * Ktot + seg;
    const unsigned short* Ag1 = A  + (size_t)(m0 + lr + 64) * Ktot + seg;
    const unsigned short* Bg0 = Bt + (size_t)(n0 + lr)      * Ktot + seg;
    const unsigned short* Bg1 = Bt + (size_t)(n0 + lr + 64) * Ktot + seg;
    u32x4 ra0 = *(const u32x4*)Ag0;
    u32x4 ra1 = *(const u32x4*)Ag1;
    u32x4 rb0 = *(const u32x4*)Bg0;
    u32x4 rb1 = *(const u32x4*)Bg1;
    for (int kt = 0; kt < nk; ++kt) {
        __syncthreads();
        *(u32x4*)(As + lr * 40 + seg)        = ra0;
        *(u32x4*)(As + (lr + 64) * 40 + seg) = ra1;
        *(u32x4*)(Bs + lr * 40 + seg)        = rb0;
        *(u32x4*)(Bs + (lr + 64) * 40 + seg) = rb1;
        __syncthreads();
        if (kt + 1 < nk) {
            const int ko = (kt + 1) << 5;
            ra0 = *(const u32x4*)(Ag0 + ko);
            ra1 = *(const u32x4*)(Ag1 + ko);
            rb0 = *(const u32x4*)(Bg0 + ko);
            rb1 = *(const u32x4*)(Bg1 + ko);
            if (kt + 2 < nk) {
                const int kp = (kt + 2) << 5;
                __builtin_prefetch(Ag0 + kp, 0, 1);
                __builtin_prefetch(Bg0 + kp, 0, 1);
            }
        }
        mma_step(As, Bs);
    }
#endif

    // Epilogue. C layout: VGPR v of lane L holds (row = v + 8*(L>=16), col = L&15).
    const int hi = lane >> 4;
#pragma unroll
    for (int i = 0; i < 2; ++i) {
#pragma unroll
        for (int j = 0; j < 4; ++j) {
            const int Cc = n0 + wn * 64 + j * 16 + cl;
#pragma unroll
            for (int v = 0; v < 8; ++v) {
                const int R = m0 + wm * 32 + i * 16 + v + 8 * hi;
                float val = acc[i][j][v];
                if (MODE == 0) {
                    const int which = Cc >> 10;
                    const int rem = Cc & (DDIM - 1);
                    const int h = rem >> 6, d = rem & (HDD - 1);
                    const float* bs = (which == 0) ? bias0
                                    : (which == 1) ? bias1 : bias2;
                    val += bs[h * HDD + d];
                    if (which == 0) val *= 0.125f;   // fold 1/sqrt(64) into Q
                    unsigned short* dst = (which == 0) ? outH0
                                        : (which == 1) ? outH1 : outH2;
                    const int b = R >> 11, t = R & (TT - 1);
                    dst[((size_t)(b * HH + h) * TT + t) * HDD + d] = f2bf(val);
                } else if (MODE == 1 || MODE == 3) {
                    val += bias0[Cc] + resid[(size_t)R * Ntot + Cc];
                    outF[(size_t)R * Ntot + Cc] = val;
                } else {  // MODE 2
                    val += bias0[Cc];
                    val = fmaxf(val, 0.0f);
                    outH0[(size_t)R * Ntot + Cc] = f2bf(val);
                }
            }
        }
    }
}

// ---------------------------------------------------------------------------
// Flash attention (causal), bf16 WMMA, fp32 online softmax.
// Q/K/V: [B*H, T, 64] bf16 (Q pre-scaled by 1/sqrt(HD)).
// Block = 256 threads (8 waves); each block handles 128 query rows of one
// (b,h); each wave owns 16 rows. K/V tiles are 64 wide. K tile is DMA'd by
// the TDM (rows of 32 DWORDs + 4 DWORDs pad -> 72-element LDS stride) while
// the threads transpose V by hand.
// ---------------------------------------------------------------------------
__global__ __launch_bounds__(256) void flash_attn_kernel(
    const unsigned short* __restrict__ Qb,
    const unsigned short* __restrict__ Kb,
    const unsigned short* __restrict__ Vb,
    unsigned short* __restrict__ Ob /* [B*T, D] bf16 */) {
    __shared__ __align__(16) unsigned short Qs[128 * 72];
    __shared__ __align__(16) unsigned short Ks[64 * 72];
    __shared__ __align__(16) unsigned short Vts[64 * 72];      // V^T: [d][s]
    __shared__ __align__(16) unsigned short Ps[8 * 16 * 72];   // per-wave P

    const int tid  = threadIdx.x;
    const int lane = tid & 31;
    const int wave = tid >> 5;
    const int kh = lane >> 4;
    const int cl = lane & 15;
    const int hi = kh;

    const int bh = blockIdx.y;
    const int q0 = blockIdx.x * 128;
    const size_t base = (size_t)bh * TT * HDD;

    // ---- load Q tile [128][64] ----
    for (int c = tid; c < 1024; c += 256) {
        const int r = c >> 3, sg = (c & 7) * 8;
        *(u32x4*)(Qs + r * 72 + sg) =
            *(const u32x4*)(Qb + base + (size_t)(q0 + r) * HDD + sg);
    }
    __syncthreads();

    const unsigned short* qrow = Qs + (wave * 16 + cl) * 72;
    const v16bf qf0 = load_frag(qrow, 0, kh);
    const v16bf qf1 = load_frag(qrow, 32, kh);

    v8f oacc[4];
    for (int dt = 0; dt < 4; ++dt) oacc[dt] = v8f{};
    float mst[8], lst[8];
#pragma unroll
    for (int v = 0; v < 8; ++v) { mst[v] = -3.0e38f; lst[v] = 0.0f; }

    unsigned short* Pw = Ps + wave * 16 * 72;
    const unsigned short* prow = Pw + cl * 72;

    for (int s0 = 0; s0 < q0 + 128; s0 += 64) {
        __syncthreads();   // all waves done reading previous K/V tiles
#if HAVE_TDM
        if (wave == 0)     // async K tile DMA overlaps the V transpose below
            tdm_load_tile<64>((unsigned)(uintptr_t)Ks,
                              Kb + base + (size_t)s0 * HDD, HDD, 64);
#else
        for (int c = tid; c < 512; c += 256) {
            const int r = c >> 3, sg = (c & 7) * 8;
            *(u32x4*)(Ks + r * 72 + sg) =
                *(const u32x4*)(Kb + base + (size_t)(s0 + r) * HDD + sg);
        }
#endif
        // V tile transposed into [d][s]
        {
            const int sv = tid >> 2, c0 = (tid & 3) * 16;
            const unsigned short* vsrc =
                Vb + base + (size_t)(s0 + sv) * HDD + c0;
#pragma unroll
            for (int e = 0; e < 16; ++e)
                Vts[(c0 + e) * 72 + sv] = vsrc[e];
        }
#if HAVE_TDM
        if (wave == 0) __builtin_amdgcn_s_wait_tensorcnt(0);
#endif
        __syncthreads();

        // ---- S = Q K^T (16 x 64 per wave) ----
        v8f sacc[4];
#pragma unroll
        for (int st = 0; st < 4; ++st) {
            const unsigned short* krow = Ks + (st * 16 + cl) * 72;
            v16bf kf0 = load_frag(krow, 0, kh);
            v16bf kf1 = load_frag(krow, 32, kh);
            v8f s = v8f{};
            s = wmma_bf16(qf0, kf0, s);
            s = wmma_bf16(qf1, kf1, s);
            sacc[st] = s;
        }

        // ---- causal mask + online softmax (row stats live per 16-lane half)
        float pv[4][8];
#pragma unroll
        for (int v = 0; v < 8; ++v) {
            const int tr = q0 + wave * 16 + v + 8 * hi;
            float ml = -3.0e38f;
#pragma unroll
            for (int st = 0; st < 4; ++st) {
                const int sc = s0 + st * 16 + cl;
                float sval = (sc <= tr) ? sacc[st][v] : -3.0e38f;
                pv[st][v] = sval;
                ml = fmaxf(ml, sval);
            }
            for (int off = 1; off < 16; off <<= 1)
                ml = fmaxf(ml, __shfl_xor(ml, off));
            const float mnew  = fmaxf(mst[v], ml);
            const float scale = __expf(mst[v] - mnew);
            float rsum = 0.0f;
#pragma unroll
            for (int st = 0; st < 4; ++st) {
                const float p = __expf(pv[st][v] - mnew);
                pv[st][v] = p;
                rsum += p;
            }
            for (int off = 1; off < 16; off <<= 1)
                rsum += __shfl_xor(rsum, off);
            lst[v] = lst[v] * scale + rsum;
            mst[v] = mnew;
#pragma unroll
            for (int dt = 0; dt < 4; ++dt) oacc[dt][v] *= scale;
#pragma unroll
            for (int st = 0; st < 4; ++st)
                Pw[(v + 8 * hi) * 72 + st * 16 + cl] = f2bf(pv[st][v]);
        }

        // ---- O += P V  (P round-trips through per-wave LDS) ----
        v16bf pf0 = load_frag(prow, 0, kh);
        v16bf pf1 = load_frag(prow, 32, kh);
#pragma unroll
        for (int dt = 0; dt < 4; ++dt) {
            const unsigned short* vrow = Vts + (dt * 16 + cl) * 72;
            v16bf vf0 = load_frag(vrow, 0, kh);
            v16bf vf1 = load_frag(vrow, 32, kh);
            oacc[dt] = wmma_bf16(pf0, vf0, oacc[dt]);
            oacc[dt] = wmma_bf16(pf1, vf1, oacc[dt]);
        }
    }

    // ---- epilogue: O/l -> attn buffer [B*T][D], col = h*64 + d ----
    const int b = bh / HH, h = bh % HH;
#pragma unroll
    for (int dt = 0; dt < 4; ++dt) {
#pragma unroll
        for (int v = 0; v < 8; ++v) {
            const int t = q0 + wave * 16 + v + 8 * hi;
            const float val = oacc[dt][v] / lst[v];
            Ob[((size_t)(b * TT + t)) * DDIM + h * HDD + dt * 16 + cl] =
                f2bf(val);
        }
    }
}

// ---------------------------------------------------------------------------
// LayerNorm over D=1024 (fp32 in; fp32 out for residual, bf16 out for GEMM A)
// ---------------------------------------------------------------------------
__global__ __launch_bounds__(256) void ln_kernel(
    const float* __restrict__ in, const float* __restrict__ g,
    const float* __restrict__ b, float* __restrict__ outF,
    unsigned short* __restrict__ outH) {
    const int row = blockIdx.x;
    const float* p = in + (size_t)row * DDIM;
    float s = 0.0f, s2 = 0.0f;
    for (int c = threadIdx.x; c < DDIM; c += 256) {
        const float v = p[c];
        s += v; s2 += v * v;
    }
    for (int off = 16; off; off >>= 1) {
        s  += __shfl_down(s, off);
        s2 += __shfl_down(s2, off);
    }
    __shared__ float ss[8], ss2[8];
    const int w = threadIdx.x >> 5, ln = threadIdx.x & 31;
    if (ln == 0) { ss[w] = s; ss2[w] = s2; }
    __syncthreads();
    if (w == 0) {
        s  = (ln < 8) ? ss[ln]  : 0.0f;
        s2 = (ln < 8) ? ss2[ln] : 0.0f;
        for (int off = 4; off; off >>= 1) {
            s  += __shfl_down(s, off);
            s2 += __shfl_down(s2, off);
        }
        if (ln == 0) { ss[0] = s; ss2[0] = s2; }
    }
    __syncthreads();
    const float mean = ss[0] * (1.0f / DDIM);
    const float var  = ss2[0] * (1.0f / DDIM) - mean * mean;
    const float rstd = rsqrtf(var + 1e-5f);
    for (int c = threadIdx.x; c < DDIM; c += 256) {
        const float v = (p[c] - mean) * rstd * g[c] + b[c];
        if (outF) outF[(size_t)row * DDIM + c] = v;
        if (outH) outH[(size_t)row * DDIM + c] = f2bf(v);
    }
}

// ---------------------------------------------------------------------------
// Launcher
// ---------------------------------------------------------------------------
extern "C" void kernel_launch(void* const* d_in, const int* in_sizes, int n_in,
                              void* d_out, int out_size, void* d_ws,
                              size_t ws_size, hipStream_t stream) {
    (void)in_sizes; (void)n_in; (void)out_size; (void)ws_size;
    const float* x   = (const float*)d_in[0];
    const float* Wq  = (const float*)d_in[1];
    const float* bq  = (const float*)d_in[2];
    const float* Wk  = (const float*)d_in[3];
    const float* bk  = (const float*)d_in[4];
    const float* Wv  = (const float*)d_in[5];
    const float* bv  = (const float*)d_in[6];
    const float* Wo  = (const float*)d_in[7];
    const float* bo  = (const float*)d_in[8];
    const float* g1  = (const float*)d_in[9];
    const float* be1 = (const float*)d_in[10];
    const float* W1  = (const float*)d_in[11];
    const float* bf1 = (const float*)d_in[12];
    const float* W2  = (const float*)d_in[13];
    const float* bf2 = (const float*)d_in[14];
    const float* g2  = (const float*)d_in[15];
    const float* be2 = (const float*)d_in[16];
    float* out = (float*)d_out;

    char*  ws  = (char*)d_ws;
    size_t off = 0;
    auto alloc = [&](size_t bytes) -> void* {
        void* p = ws + off;
        off = (off + bytes + 255) & ~(size_t)255;
        return p;
    };
    unsigned short* xb     = (unsigned short*)alloc((size_t)MROWS * DDIM * 2);
    unsigned short* Wqkvt  = (unsigned short*)alloc((size_t)3 * DDIM * DDIM * 2);
    unsigned short* Wot    = (unsigned short*)alloc((size_t)DDIM * DDIM * 2);
    unsigned short* W1t    = (unsigned short*)alloc((size_t)FFD * DDIM * 2);
    unsigned short* W2t    = (unsigned short*)alloc((size_t)DDIM * FFD * 2);
    unsigned short* qbuf   = (unsigned short*)alloc((size_t)BB * HH * TT * HDD * 2);
    unsigned short* kbuf   = (unsigned short*)alloc((size_t)BB * HH * TT * HDD * 2);
    unsigned short* vbuf   = (unsigned short*)alloc((size_t)BB * HH * TT * HDD * 2);
    unsigned short* attnb  = (unsigned short*)alloc((size_t)MROWS * DDIM * 2);
    float*          res    = (float*)alloc((size_t)MROWS * DDIM * 4);
    float*          out1F  = (float*)alloc((size_t)MROWS * DDIM * 4);
    unsigned short* out1H  = (unsigned short*)alloc((size_t)MROWS * DDIM * 2);
    unsigned short* hidden = (unsigned short*)alloc((size_t)MROWS * FFD * 2);

    // --- prep ---
    cvt_bf16_kernel<<<4096, 256, 0, stream>>>(x, xb, (size_t)MROWS * DDIM);
    pack_qkv_w_kernel<<<4096, 256, 0, stream>>>(Wq, Wk, Wv, Wqkvt);
    transpose_bf16_kernel<<<4096, 256, 0, stream>>>(Wo, Wot, DDIM, DDIM);
    transpose_bf16_kernel<<<8192, 256, 0, stream>>>(W1, W1t, DDIM, FFD);
    transpose_bf16_kernel<<<8192, 256, 0, stream>>>(W2, W2t, FFD, DDIM);

    // --- QKV projection: [8192,1024] x [1024,3072] -> q/k/v [B,H,T,64] ---
    gemm_bf16_kernel<0><<<dim3(3 * DDIM / 128, MROWS / 128), 256, 0, stream>>>(
        xb, Wqkvt, 3 * DDIM, DDIM, bq, bk, bv, nullptr, nullptr,
        qbuf, kbuf, vbuf);

    // --- causal flash attention ---
    flash_attn_kernel<<<dim3(TT / 128, BB * HH), 256, 0, stream>>>(
        qbuf, kbuf, vbuf, attnb);

    // --- output projection + residual: res = x + attn @ Wo + bo ---
    gemm_bf16_kernel<1><<<dim3(DDIM / 128, MROWS / 128), 256, 0, stream>>>(
        attnb, Wot, DDIM, DDIM, bo, nullptr, nullptr, x, res,
        nullptr, nullptr, nullptr);

    // --- LN1 ---
    ln_kernel<<<MROWS, 256, 0, stream>>>(res, g1, be1, out1F, out1H);

    // --- FFN1: hidden = relu(out1 @ W1 + b1), bf16 [8192,16384] ---
    gemm_bf16_kernel<2><<<dim3(FFD / 128, MROWS / 128), 256, 0, stream>>>(
        out1H, W1t, FFD, DDIM, bf1, nullptr, nullptr, nullptr, nullptr,
        hidden, nullptr, nullptr);

    // --- FFN2 + residual: res = out1 + hidden @ W2 + b2 ---
    gemm_bf16_kernel<3><<<dim3(DDIM / 128, MROWS / 128), 256, 0, stream>>>(
        hidden, W2t, DDIM, FFD, bf2, nullptr, nullptr, out1F, res,
        nullptr, nullptr, nullptr);

    // --- LN2 -> final output ---
    ln_kernel<<<MROWS, 256, 0, stream>>>(res, g2, be2, out, nullptr);
}
